// EMMA_43061342110089
// MI455X (gfx1250) — matrix operand order repaired and checked by reading, not compiled, not run on hardware
//
#include <hip/hip_runtime.h>

typedef __attribute__((ext_vector_type(16))) _Float16 v16h;
typedef __attribute__((ext_vector_type(8)))  float    v8f;
typedef __attribute__((ext_vector_type(4)))  unsigned u32x4;
typedef __attribute__((ext_vector_type(8)))  unsigned u32x8;

#define NCELL 10000
#define EMB   256
#define TXT   768
#define S16   16
#define T36   36
#define FM    64
#define KTOT  1024     // 256*2*2
#define NPOS  9801     // 99*99
#define NLAT  128
#define ADIM  5
#define VOCAB 25
#define LIN   627264   // 64*99*99

__device__ __forceinline__ float lrelu(float x) { return x > 0.f ? x : 0.01f * x; }

// ---- WMMA fragment loaders (CDNA5 wave32 16-bit layouts, CDNA5 ISA) ----
// A (16x32 f16): both lane halves hold M=lane&15; VGPR v -> K=h*16 + v*2 + j
__device__ __forceinline__ v16h load_a16(const _Float16* __restrict__ A, int lda,
                                         int row0, int k0, int lane) {
  const _Float16* base = A + (size_t)(row0 + (lane & 15)) * lda + k0;
  int h = lane >> 4;
  v16h a;
#pragma unroll
  for (int v = 0; v < 8; ++v) {
    int k = h * 16 + v * 2;
    a[2 * v]     = base[k];
    a[2 * v + 1] = base[k + 1];
  }
  return a;
}
// B (32x16 f16): N = lane&15; lanes 0-15 K=0..15, lanes 16-31 K=16..31; K = h*16 + 2v + j
__device__ __forceinline__ v16h load_b16(const _Float16* __restrict__ B, int ldb,
                                         int k0, int n0, int lane) {
  int n = n0 + (lane & 15);
  int h = lane >> 4;
  v16h b;
#pragma unroll
  for (int v = 0; v < 8; ++v) {
    int k = k0 + h * 16 + 2 * v;
    b[2 * v]     = B[(size_t)k * ldb + n];
    b[2 * v + 1] = B[(size_t)(k + 1) * ldb + n];
  }
  return b;
}

// ---- K0: per-vocab nonzero flags + zero the W1 accumulator ----
__global__ void k0_init(const float* __restrict__ semb, int* __restrict__ nz,
                        float* __restrict__ h1acc) {
  int b = blockIdx.x, t = threadIdx.x;
  if (b < VOCAB) {
    __shared__ int flag;
    if (t == 0) flag = 0;
    __syncthreads();
    if (semb[b * EMB + t] != 0.f) atomicOr(&flag, 1);
    __syncthreads();
    if (t == 0) nz[b] = flag;
  } else {
    if (t < NLAT) h1acc[t] = 0.f;
  }
}

// ---- K1: token-softmax scales and scale-weighted token sums (per sprite s) ----
__global__ void __launch_bounds__(256)
k1_toksum(const float* __restrict__ temb, const float* __restrict__ Wsk,
          const float* __restrict__ bsk, const float* __restrict__ Wsv,
          const float* __restrict__ bsv, float* __restrict__ wsumK,
          float* __restrict__ wsumV) {
  int s = blockIdx.x, tid = threadIdx.x;
  int lane = tid & 31, wv = tid >> 5;
  __shared__ float lk[T36], lv[T36], sck[T36], scv[T36];
  const float* tb = temb + (size_t)s * T36 * TXT;
  for (int t = wv; t < T36; t += 8) {
    float pk = 0.f, pv = 0.f;
    for (int c = lane; c < TXT; c += 32) {
      float x = tb[t * TXT + c];
      pk += x * Wsk[c];
      pv += x * Wsv[c];
    }
    for (int m = 16; m >= 1; m >>= 1) {
      pk += __shfl_xor(pk, m, 32);
      pv += __shfl_xor(pv, m, 32);
    }
    if (lane == 0) { lk[t] = pk + bsk[0]; lv[t] = pv + bsv[0]; }
  }
  __syncthreads();
  if (tid == 0) {
    float mk = -1e30f, mv = -1e30f;
    for (int t = 0; t < T36; ++t) { mk = fmaxf(mk, lk[t]); mv = fmaxf(mv, lv[t]); }
    float sk = 0.f, sv = 0.f;
    for (int t = 0; t < T36; ++t) {
      sck[t] = __expf(lk[t] - mk); sk += sck[t];
      scv[t] = __expf(lv[t] - mv); sv += scv[t];
    }
    for (int t = 0; t < T36; ++t) { sck[t] /= sk; scv[t] /= sv; }
  }
  __syncthreads();
  for (int e = tid; e < TXT; e += 256) {
    float ak = 0.f, av = 0.f;
    for (int t = 0; t < T36; ++t) {
      float x = tb[t * TXT + e];
      ak += sck[t] * x;
      av += scv[t] * x;
    }
    wsumK[s * TXT + e] = ak;
    wsumV[s * TXT + e] = av;
  }
}

// ---- K2: key = wsumK @ Wk + bk (keyT f16 [EMB][S]); value = wsumV @ Wv + bv ----
__global__ void __launch_bounds__(256)
k2_keyval(const float* __restrict__ wsumK, const float* __restrict__ wsumV,
          const float* __restrict__ Wk, const float* __restrict__ bk,
          const float* __restrict__ Wv, const float* __restrict__ bv,
          _Float16* __restrict__ keyT, float* __restrict__ value) {
  int s = blockIdx.x, e = threadIdx.x;
  float ak = bk[e], av = bv[e];
  for (int c = 0; c < TXT; ++c) {
    ak += wsumK[s * TXT + c] * Wk[c * EMB + e];
    av += wsumV[s * TXT + c] * Wv[c * EMB + e];
  }
  value[s * EMB + e] = av;
  keyT[e * S16 + s] = (_Float16)ak;
}

// ---- K3: sprite gather with nonzero-mean -> query(f16) and avatar_emb(f32) ----
__global__ void __launch_bounds__(256)
k3_gather(const int* __restrict__ ent, const int* __restrict__ av,
          const float* __restrict__ semb, const int* __restrict__ nz,
          _Float16* __restrict__ qf16, float* __restrict__ avemb) {
  int cell = blockIdx.x, e = threadIdx.x;
  int i0 = ent[cell * 4 + 0], i1 = ent[cell * 4 + 1];
  int i2 = ent[cell * 4 + 2], i3 = ent[cell * 4 + 3];
  int cnt = nz[i0] + nz[i1] + nz[i2] + nz[i3];
  float inv = 1.f / (float)(cnt > 0 ? cnt : 1);
  float q = (semb[i0 * EMB + e] + semb[i1 * EMB + e] +
             semb[i2 * EMB + e] + semb[i3 * EMB + e]) * inv;
  qf16[(size_t)cell * EMB + e] = (_Float16)q;
  avemb[(size_t)cell * EMB + e] = semb[av[cell] * EMB + e];
}

// ---- Kw: conv weights f32 -> f16 (A matrix [64][1024]) ----
__global__ void kw_conv16(const float* __restrict__ cw, _Float16* __restrict__ cwh) {
  int i = blockIdx.x * 256 + threadIdx.x;
  if (i < FM * KTOT) cwh[i] = (_Float16)cw[i];
}

// ---- K5: WMMA kq + in-register masked softmax + weights@value + avatar mix ----
__global__ void __launch_bounds__(32)
k5_attn(const _Float16* __restrict__ qf16, const _Float16* __restrict__ keyT,
        const float* __restrict__ value, const float* __restrict__ avemb,
        _Float16* __restrict__ xin) {
  int lane = threadIdx.x;
  int row0 = blockIdx.x * 16;
  __shared__ float vls[S16 * EMB];
  __shared__ float wls[16 * 16];
  for (int i = lane; i < S16 * EMB; i += 32) vls[i] = value[i];
  v8f acc = {};
  for (int kt = 0; kt < EMB / 32; ++kt) {
    v16h a = load_a16(qf16, EMB, row0, kt * 32, lane);
    v16h b = load_b16(keyT, S16, kt * 32, 0, lane);
    acc = __builtin_amdgcn_wmma_f32_16x16x32_f16(false, a, false, b,
                                                 (short)0, acc, false, false);
  }
  int h = lane >> 4;
#pragma unroll
  for (int r = 0; r < 8; ++r) {                 // D row r+8h lives in this 16-lane half
    float kq  = acc[r];
    float msk = (kq != 0.f) ? 1.f : 0.f;        // mask pre-scale, as in reference
    float v   = kq * 0.0625f;                   // / sqrt(256)
    float mx = v;
    for (int m = 8; m >= 1; m >>= 1) mx = fmaxf(mx, __shfl_xor(mx, m, 32));
    float ex = __expf(v - mx);
    float sm = ex;
    for (int m = 8; m >= 1; m >>= 1) sm += __shfl_xor(sm, m, 32);
    wls[(r + 8 * h) * 16 + (lane & 15)] = (ex / sm) * msk;
  }
  __syncthreads();
  int r = lane & 15, ep = lane >> 4;
  int cell = row0 + r;
  for (int e = ep; e < EMB; e += 2) {
    float o = 0.f;
#pragma unroll
    for (int s = 0; s < S16; ++s) o += wls[r * 16 + s] * vls[s * EMB + e];
    float out = (o * 0.0625f + avemb[(size_t)cell * EMB + e]) * 0.5f;
    xin[(size_t)e * NCELL + cell] = (_Float16)out;   // [C][H*W] layout for conv
  }
}

// ============================================================================
// K6: conv as implicit GEMM. One block = one output row (oy): M=64 filters x
// N=112 (7 n-tiles of 16 covering ox 0..98). Per K-step (8 channels x 2x2):
//   * weight tile 64x32 f16 -> LDS via TENSOR_LOAD_TO_LDS (TDM, TENSORcnt)
//   * input slab 8ch x 2rows x 100cols -> LDS via GLOBAL_LOAD_ASYNC_TO_LDS_B64
// then 8 waves run 28 (M,N) WMMA tiles against the shared LDS data.
// ============================================================================
__device__ __forceinline__ v16h lds_a_frag(const _Float16* sw, int row0, int lane) {
  const _Float16* base = sw + (row0 + (lane & 15)) * 32;
  int h = lane >> 4;
  v16h a;
#pragma unroll
  for (int v = 0; v < 8; ++v) {
    int k = h * 16 + v * 2;
    a[2 * v]     = base[k];
    a[2 * v + 1] = base[k + 1];
  }
  return a;
}
__device__ __forceinline__ v16h lds_b_frag(const _Float16* sx, int n0, int lane) {
  int n   = n0 + (lane & 15);
  int ncl = n < 99 ? n : 98;                       // clamp, branch-free
  _Float16 vf = (_Float16)(n < 99 ? 1.0f : 0.0f);  // zero pad lanes
  int h = lane >> 4;
  v16h b;
#pragma unroll
  for (int v = 0; v < 8; ++v) {
#pragma unroll
    for (int j = 0; j < 2; ++j) {
      int kk = h * 16 + 2 * v + j;              // local k 0..31
      int c = kk >> 2, ky = (kk >> 1) & 1, kx = kk & 1;
      b[2 * v + j] = sx[(c * 2 + ky) * 100 + ncl + kx] * vf;
    }
  }
  return b;
}

__global__ void __launch_bounds__(256)
k6_conv(const _Float16* __restrict__ cwh, const _Float16* __restrict__ xin,
        const float* __restrict__ cb, float* __restrict__ xact) {
  __shared__ __align__(16) _Float16 sx[8 * 2 * 100];  // input slab  [c][ky][col]
  __shared__ __align__(16) _Float16 sw[64 * 32];      // weight tile [m][k]
  int tid = threadIdx.x, lane = tid & 31, wv = tid >> 5;
  int oy = blockIdx.x;

  v8f acc[4] = {{}, {}, {}, {}};

  for (int kt = 0; kt < 32; ++kt) {
    __syncthreads();   // previous iteration's LDS readers done before restage

    // --- TDM: weight tile cwh[0:64, kt*32 : kt*32+32] -> sw (issued by wave 0) ---
    if (wv == 0) {
      unsigned long long ga = (unsigned long long)(const void*)cwh
                            + (unsigned long long)(kt * 32 * 2);
      unsigned lw = (unsigned)(size_t)(const void*)sw;   // LDS byte addr (aperture low 32)
      u32x4 g0;
      g0[0] = 1u;                                  // count=1, user mode
      g0[1] = lw;                                  // lds_addr
      g0[2] = (unsigned)(ga & 0xffffffffu);        // global_addr[31:0]
      g0[3] = (unsigned)(ga >> 32) | (2u << 30);   // global_addr[56:32] | type=2
      u32x8 g1;
      g1[0] = 1u << 16;          // data_size=1 (2 bytes)
      g1[1] = 1024u << 16;       // tensor_dim0[15:0]=1024 @bits63:48
      g1[2] = 64u << 16;         // tensor_dim0[31:16]=0 | tensor_dim1[15:0]=64
      g1[3] = 32u << 16;         // tensor_dim1[31:16]=0 | tile_dim0=32
      g1[4] = 64u;               // tile_dim1=64, tile_dim2=0
      g1[5] = 1024u;             // tensor_dim0_stride[31:0]=1024
      g1[6] = 0u;                // stride0[47:32]=0 | stride1[15:0]=0
      g1[7] = 0u;
      asm volatile("tensor_load_to_lds %0, %1" :: "s"(g0), "s"(g1) : "memory");
    }

    // --- async stage: 8 channels x 2 rows x 100 cols = 400 qwords -> sx ---
    int c0 = kt * 8;
#pragma unroll
    for (int rr = 0; rr < 2; ++rr) {
      int q = tid + rr * 256;
      if (q < 400) {
        int c  = q / 50, rm = q % 50;
        int ky = rm / 25, qw = rm % 25;
        int col = qw * 4;
        const _Float16* gp = xin + (size_t)(c0 + c) * NCELL + (oy + ky) * 100 + col;
        unsigned la = (unsigned)(size_t)(const void*)(sx + (c * 2 + ky) * 100 + col);
        unsigned long long gaq = (unsigned long long)(const void*)gp;
        asm volatile("global_load_async_to_lds_b64 %0, %1, off"
                     :: "v"(la), "v"(gaq) : "memory");
      }
    }
    asm volatile("s_wait_asynccnt 0x0" ::: "memory");
    if (wv == 0) __builtin_amdgcn_s_wait_tensorcnt(0);
    __syncthreads();   // all waves' staged data visible

    // --- compute: wave wv covers (M,N) pairs p = wv, wv+8, wv+16, wv+24 ---
#pragma unroll
    for (int i = 0; i < 4; ++i) {
      int p = wv + 8 * i;
      if (p < 28) {
        int m = p & 3, nt = p >> 2;
        v16h a = lds_a_frag(sw, m * 16, lane);
        v16h b = lds_b_frag(sx, nt * 16, lane);
        acc[i] = __builtin_amdgcn_wmma_f32_16x16x32_f16(false, a, false, b,
                                                        (short)0, acc[i], false, false);
      }
    }
  }

  // --- epilogue: bias + leaky-relu, flatten order f*9801 + oy*99 + ox ---
  int h = lane >> 4;
#pragma unroll
  for (int i = 0; i < 4; ++i) {
    int p = wv + 8 * i;
    if (p < 28) {
      int m = p & 3, nt = p >> 2;
      int n = nt * 16 + (lane & 15);
      if (n < 99) {
#pragma unroll
        for (int r = 0; r < 8; ++r) {
          int M = m * 16 + r + 8 * h;
          xact[(size_t)M * NPOS + oy * 99 + n] = lrelu(acc[i][r] + cb[M]);
        }
      }
    }
  }
}

// ---- K7: x @ W1 — the 321 MB bandwidth-bound matvec, split-K with atomics ----
__global__ void __launch_bounds__(256)
k7_w1(const float* __restrict__ xact, const float* __restrict__ W1,
      float* __restrict__ h1acc) {
  __shared__ float accl[NLAT];
  int tid = threadIdx.x, lane = tid & 31, wv = tid >> 5;
  if (tid < NLAT) accl[tid] = 0.f;
  __syncthreads();
  int gw = blockIdx.x * 8 + wv;
  int nw = gridDim.x * 8;
  float a0 = 0.f, a1 = 0.f, a2 = 0.f, a3 = 0.f;
  for (int i = gw; i < LIN; i += nw) {
    __builtin_prefetch(W1 + (size_t)(i + nw) * NLAT + lane * 4, 0, 1);
    float xv = xact[i];                                  // uniform -> scalar load
    const float4 w4 = *(const float4*)(W1 + (size_t)i * NLAT + lane * 4);
    a0 += xv * w4.x; a1 += xv * w4.y; a2 += xv * w4.z; a3 += xv * w4.w;
  }
  atomicAdd(&accl[lane * 4 + 0], a0);
  atomicAdd(&accl[lane * 4 + 1], a1);
  atomicAdd(&accl[lane * 4 + 2], a2);
  atomicAdd(&accl[lane * 4 + 3], a3);
  __syncthreads();
  if (tid < NLAT) atomicAdd(&h1acc[tid], accl[tid]);
}

// ---- K8: MLP head + softmax ----
__global__ void __launch_bounds__(128)
k8_head(const float* __restrict__ h1acc, const float* __restrict__ b1,
        const float* __restrict__ Wh, const float* __restrict__ bh,
        const float* __restrict__ Wo, const float* __restrict__ bo,
        float* __restrict__ out) {
  __shared__ float h1[NLAT], h2[NLAT], lg[ADIM], red[2];
  int t = threadIdx.x;
  h1[t] = lrelu(h1acc[t] + b1[t]);
  __syncthreads();
  float s = bh[t];
  for (int c = 0; c < NLAT; ++c) s += h1[c] * Wh[c * NLAT + t];
  h2[t] = lrelu(s);
  __syncthreads();
  if (t < ADIM) {
    float l = bo[t];
    for (int c = 0; c < NLAT; ++c) l += h2[c] * Wo[c * ADIM + t];
    lg[t] = l;
  }
  __syncthreads();
  if (t == 0) {
    float m = lg[0];
    for (int a = 1; a < ADIM; ++a) m = fmaxf(m, lg[a]);
    float sm = 0.f;
    for (int a = 0; a < ADIM; ++a) sm += __expf(lg[a] - m);
    red[0] = m; red[1] = sm;
  }
  __syncthreads();
  if (t < ADIM) out[t] = __expf(lg[t] - red[0]) / red[1];
}

extern "C" void kernel_launch(void* const* d_in, const int* in_sizes, int n_in,
                              void* d_out, int out_size, void* d_ws, size_t ws_size,
                              hipStream_t stream) {
  (void)in_sizes; (void)n_in; (void)out_size; (void)ws_size;
  const int*   entities = (const int*)d_in[0];
  const int*   avatar   = (const int*)d_in[1];
  const float* temb     = (const float*)d_in[2];
  const float* semb     = (const float*)d_in[3];
  const float* Wk  = (const float*)d_in[4];
  const float* bk  = (const float*)d_in[5];
  const float* Wsk = (const float*)d_in[6];
  const float* bsk = (const float*)d_in[7];
  const float* Wv  = (const float*)d_in[8];
  const float* bv  = (const float*)d_in[9];
  const float* Wsv = (const float*)d_in[10];
  const float* bsv = (const float*)d_in[11];
  const float* convw = (const float*)d_in[12];
  const float* convb = (const float*)d_in[13];
  const float* W1 = (const float*)d_in[14];
  const float* b1 = (const float*)d_in[15];
  const float* Wh = (const float*)d_in[16];
  const float* bh = (const float*)d_in[17];
  const float* Wo = (const float*)d_in[18];
  const float* bo = (const float*)d_in[19];
  float* out = (float*)d_out;

  char* ws = (char*)d_ws;
  size_t off = 0;
  auto take = [&](size_t bytes) -> char* {
    char* p = ws + off;
    off = (off + bytes + 255) & ~(size_t)255;
    return p;
  };
  int*      nz    = (int*)take(32 * sizeof(int));
  float*    h1acc = (float*)take(NLAT * sizeof(float));
  float*    wsumK = (float*)take((size_t)S16 * TXT * sizeof(float));
  float*    wsumV = (float*)take((size_t)S16 * TXT * sizeof(float));
  float*    value = (float*)take((size_t)S16 * EMB * sizeof(float));
  _Float16* keyT  = (_Float16*)take((size_t)EMB * S16 * sizeof(_Float16));
  _Float16* cwh   = (_Float16*)take((size_t)FM * KTOT * sizeof(_Float16));
  _Float16* qf16  = (_Float16*)take((size_t)NCELL * EMB * sizeof(_Float16));
  float*    avemb = (float*)take((size_t)NCELL * EMB * sizeof(float));
  _Float16* xin   = (_Float16*)take((size_t)EMB * NCELL * sizeof(_Float16));
  float*    xact  = (float*)take((size_t)LIN * sizeof(float));

  k0_init<<<VOCAB + 1, 256, 0, stream>>>(semb, nz, h1acc);
  k1_toksum<<<S16, 256, 0, stream>>>(temb, Wsk, bsk, Wsv, bsv, wsumK, wsumV);
  k2_keyval<<<S16, 256, 0, stream>>>(wsumK, wsumV, Wk, bk, Wv, bv, keyT, value);
  k3_gather<<<NCELL, 256, 0, stream>>>(entities, avatar, semb, nz, qf16, avemb);
  kw_conv16<<<(FM * KTOT + 255) / 256, 256, 0, stream>>>(convw, cwh);
  k5_attn<<<NCELL / 16, 32, 0, stream>>>(qf16, keyT, value, avemb, xin);
  k6_conv<<<99, 256, 0, stream>>>(cwh, xin, convb, xact);
  k7_w1<<<1024, 256, 0, stream>>>(xact, W1, h1acc);
  k8_head<<<1, 128, 0, stream>>>(h1acc, b1, Wh, bh, Wo, bo, out);
}
